// RANet_40553081209538
// MI455X (gfx1250) — matrix operand order repaired and checked by reading, not compiled, not run on hardware
//
#include <hip/hip_runtime.h>
#include <hip/hip_bf16.h>
#include <math.h>

#define N_NODES 32768
#define F_IN    128
#define DMODEL  512
#define NDOCS   256
#define NEDGES  262144

typedef __attribute__((ext_vector_type(16))) _Float16 v16h;
typedef __attribute__((ext_vector_type(8)))  _Float16 v8h;
typedef __attribute__((ext_vector_type(4)))  _Float16 v4h;
typedef __attribute__((ext_vector_type(8)))  float    v8f;
typedef __attribute__((ext_vector_type(4)))  float    v4f;

// ---------------------------------------------------------------------------
// Degree / normalization
// ---------------------------------------------------------------------------
__global__ void deg_count(const int* __restrict__ dst, float* __restrict__ deg, int E) {
    int e = blockIdx.x * 256 + threadIdx.x;
    if (e < E) atomicAdd(&deg[dst[e]], 1.0f);
}

__global__ void make_dis(const float* __restrict__ deg, float* __restrict__ dis, int n) {
    int i = blockIdx.x * 256 + threadIdx.x;
    if (i < n) dis[i] = rsqrtf(deg[i] + 1.0f);
}

// ---------------------------------------------------------------------------
// Attention: per-doc mean, partner lookup (d ^ 1)
// ---------------------------------------------------------------------------
__global__ void attn_acc(const int* __restrict__ doc, const float* __restrict__ att,
                         float* __restrict__ s, float* __restrict__ c, int n) {
    int i = blockIdx.x * 256 + threadIdx.x;
    if (i < n) {
        int d = doc[i];
        atomicAdd(&s[d], att[i]);
        atomicAdd(&c[d], 1.0f);
    }
}

__global__ void attn_a(const float* __restrict__ s, const float* __restrict__ c,
                       float* __restrict__ a) {
    int d = threadIdx.x;           // NDOCS == 256 == blockDim
    int p = d ^ 1;
    a[d] = s[p] / c[p];
}

// ---------------------------------------------------------------------------
// fp32 -> f16 conversion, 4-wide (input features)
// ---------------------------------------------------------------------------
__global__ void cvt_f16(const float* __restrict__ x, _Float16* __restrict__ o, int n4) {
    int i = blockIdx.x * 256 + threadIdx.x;
    if (i >= n4) return;
    v4f v = *(const v4f*)(x + (long)i * 4);
    v4h h = { (_Float16)v.x, (_Float16)v.y, (_Float16)v.z, (_Float16)v.w };
    *(v4h*)(o + (long)i * 4) = h;
}

// ---------------------------------------------------------------------------
// Weight pack: fp32 [K x Dout] row-major -> f16 WMMA-B fragment-major
// layout: [ntile][kstep][lane 0..31][16 halves]
//   lane L holds column ntile*16 + (L&15), K range (L>>4)*16 .. +15
// ---------------------------------------------------------------------------
__global__ void pack_w(const float* __restrict__ W, _Float16* __restrict__ P,
                       int K, int Dout) {
    int idx = blockIdx.x * 256 + threadIdx.x;
    if (idx >= K * Dout) return;
    int j      = idx & 15;
    int lane   = (idx >> 4) & 31;
    int rest   = idx >> 9;                 // ntile*ksteps + ks
    int ksteps = K >> 5;
    int ks     = rest % ksteps;
    int ntile  = rest / ksteps;
    int srcK   = ks * 32 + (lane >> 4) * 16 + j;
    int col    = ntile * 16 + (lane & 15);
    P[idx] = (_Float16)W[(long)srcK * Dout + col];
}

// ---------------------------------------------------------------------------
// WMMA GEMM helpers: ping-pong fragment buffers, no copy on swap
// ---------------------------------------------------------------------------
union AFrag { v16h v; v8h h[2]; };

__device__ __forceinline__ void load_a(AFrag& a, const _Float16* aPtr, int ks) {
    a.h[0] = *(const v8h*)(aPtr + ks * 32);
    a.h[1] = *(const v8h*)(aPtr + ks * 32 + 8);
}

template <int NTG>
__device__ __forceinline__ void load_b(v16h (&b)[NTG], const _Float16* bPtr,
                                       long ks, long bTile) {
#pragma unroll
    for (int t = 0; t < NTG; ++t)
        b[t] = *(const v16h*)(bPtr + ks * (32 * 16) + t * bTile);
}

template <int NTG>
__device__ __forceinline__ void do_wmma(v8f (&acc)[NTG], const AFrag& a,
                                        const v16h (&b)[NTG]) {
#pragma unroll
    for (int t = 0; t < NTG; ++t)
        acc[t] = __builtin_amdgcn_wmma_f32_16x16x32_f16(
            false, a.v, false, b[t], (short)0, acc[t], false, false);
}

// ---------------------------------------------------------------------------
// WMMA GEMM: C[N x Dout] = A[N x K] * B  (A f16 row-major, B fragment-packed)
// One wave: 16-row strip x NTG consecutive 16-col tiles; k-loop unrolled x2
// with fixed ping-pong buffers so loads overlap WMMAs with no register copies.
// mode 0 (GCN): Cout = f16( acc * dis[row] )          -> y = dis ⊙ (hW)
// mode 1 (FC) : Cout = f16( tanh(acc + bias[col]) )
// ---------------------------------------------------------------------------
template <int NTG>
__global__ __launch_bounds__(256)
void wmma_gemm(const _Float16* __restrict__ A, const _Float16* __restrict__ Bpk,
               _Float16* __restrict__ Cout, const int K, const int Dout,
               const int mode, const float* __restrict__ dis,
               const float* __restrict__ bias) {
    const int lane    = threadIdx.x & 31;
    const int wave    = (blockIdx.x << 3) + (threadIdx.x >> 5);
    const int ngroups = (Dout >> 4) / NTG;
    const int mtile   = wave / ngroups;
    const int ng      = wave % ngroups;
    const int hi      = lane >> 4;
    const int lo      = lane & 15;
    const int ksteps  = K >> 5;            // even: 4 (K=128) or 16 (K=512)

    // A fragment (16x32 f16): lanes 0-15 hold K 0-15, lanes 16-31 K 16-31
    const _Float16* aPtr = A + (long)(mtile * 16 + lo) * K + hi * 16;
    // B fragment: one contiguous 32B load per lane from packed weights
    const _Float16* bPtr = Bpk + (((long)(ng * NTG) * ksteps) * 32 + lane) * 16;
    const long bTile = (long)ksteps * 32 * 16;     // halves between ntiles

    v8f acc[NTG] = {};
    AFrag a0, a1;
    v16h  b0[NTG], b1[NTG];

    load_a(a0, aPtr, 0);  load_b<NTG>(b0, bPtr, 0, bTile);
    load_a(a1, aPtr, 1);  load_b<NTG>(b1, bPtr, 1, bTile);

    for (int ks = 0; ks < ksteps - 2; ks += 2) {
        do_wmma<NTG>(acc, a0, b0);
        load_a(a0, aPtr, ks + 2);  load_b<NTG>(b0, bPtr, ks + 2, bTile);
        do_wmma<NTG>(acc, a1, b1);
        load_a(a1, aPtr, ks + 3);  load_b<NTG>(b1, bPtr, ks + 3, bTile);
    }
    do_wmma<NTG>(acc, a0, b0);
    do_wmma<NTG>(acc, a1, b1);

    // C layout: VGPR r -> row mtile*16 + hi*8 + r, col ntile*16 + lo
#pragma unroll
    for (int t = 0; t < NTG; ++t) {
        const int col = (ng * NTG + t) * 16 + lo;
        if (mode == 0) {
#pragma unroll
            for (int r = 0; r < 8; ++r) {
                const int row = mtile * 16 + hi * 8 + r;
                Cout[(long)row * Dout + col] = (_Float16)(acc[t][r] * dis[row]);
            }
        } else {
            const float bb = bias[col];
#pragma unroll
            for (int r = 0; r < 8; ++r) {
                const int row = mtile * 16 + hi * 8 + r;
                Cout[(long)row * Dout + col] = (_Float16)tanhf(acc[t][r] + bb);
            }
        }
    }
}

// ---------------------------------------------------------------------------
// Edge scatter: agg[dst] += y[src]  (y f16, agg fp32, D=512)
// 64 threads per edge, 8 halves (16B load) per thread, 4 edges per block
// ---------------------------------------------------------------------------
__global__ __launch_bounds__(256)
void edge_scatter(const int* __restrict__ src, const int* __restrict__ dst,
                  const _Float16* __restrict__ y, float* __restrict__ agg, int E) {
    int e = (blockIdx.x << 2) + (threadIdx.x >> 6);
    if (e >= E) return;
    int t = threadIdx.x & 63;
    int s = src[e];
    int d = dst[e];
    const v8h v = *(const v8h*)(y + (long)s * DMODEL + t * 8);
    float* ap = agg + (long)d * DMODEL + t * 8;
#pragma unroll
    for (int j = 0; j < 8; ++j) atomicAdd(ap + j, (float)v[j]);
}

// ---------------------------------------------------------------------------
// GCN epilogue, 4-wide: h = tanh( dis[row] * (agg + y) + b )   (y = dis ⊙ hW)
// ---------------------------------------------------------------------------
__global__ void gcn_combine(const float* __restrict__ agg, const _Float16* __restrict__ y,
                            const float* __restrict__ dis, const float* __restrict__ bias,
                            _Float16* __restrict__ hout, int n4) {
    int i = blockIdx.x * 256 + threadIdx.x;       // index of 4-element group
    if (i >= n4) return;
    long base = (long)i * 4;
    int row = (int)(base >> 9);                   // D = 512
    int col = (int)(base & 511);
    float dr = dis[row];
    v4f s = *(const v4f*)(agg + base);
    v4h yv = *(const v4h*)(y + base);
    v4f bb = *(const v4f*)(bias + col);
    v4h o;
    o.x = (_Float16)tanhf(dr * (s.x + (float)yv.x) + bb.x);
    o.y = (_Float16)tanhf(dr * (s.y + (float)yv.y) + bb.y);
    o.z = (_Float16)tanhf(dr * (s.z + (float)yv.z) + bb.z);
    o.w = (_Float16)tanhf(dr * (s.w + (float)yv.w) + bb.w);
    *(v4h*)(hout + base) = o;
}

// ---------------------------------------------------------------------------
// fc3: [N,32] @ [32,1] + b
// ---------------------------------------------------------------------------
__global__ void fc3_kernel(const _Float16* __restrict__ h2, const float* __restrict__ W3,
                           const float* __restrict__ b3, float* __restrict__ h3, int n) {
    int i = blockIdx.x * 256 + threadIdx.x;
    if (i >= n) return;
    const _Float16* p = h2 + (long)i * 32;
    float acc = b3[0];
#pragma unroll
    for (int j = 0; j < 32; ++j) acc += (float)p[j] * W3[j];
    h3[i] = acc;
}

// ---------------------------------------------------------------------------
// Mean pool + preference sigmoid
// ---------------------------------------------------------------------------
__global__ void pool_acc(const int* __restrict__ batch, const float* __restrict__ h3,
                         float* __restrict__ s, float* __restrict__ c, int n) {
    int i = blockIdx.x * 256 + threadIdx.x;
    if (i < n) {
        int d = batch[i];
        atomicAdd(&s[d], h3[i]);
        atomicAdd(&c[d], 1.0f);
    }
}

__global__ void final_kernel(const float* __restrict__ s, const float* __restrict__ c,
                             const float* __restrict__ a, const int* __restrict__ ia,
                             const int* __restrict__ ib, float* __restrict__ out) {
    int d = threadIdx.x;          // NDOCS == 256
    int ja = ia[d], jb = ib[d];
    float ua = s[ja] / c[ja];
    float ub = s[jb] / c[jb];
    float z  = (ub - ua) * a[d];
    out[d] = 1.0f / (1.0f + expf(-z));
}

// ---------------------------------------------------------------------------
extern "C" void kernel_launch(void* const* d_in, const int* in_sizes, int n_in,
                              void* d_out, int out_size, void* d_ws, size_t ws_size,
                              hipStream_t stream) {
    (void)in_sizes; (void)n_in; (void)out_size; (void)ws_size;

    const float* x     = (const float*)d_in[0];
    const int*   eidx  = (const int*)d_in[1];
    const int*   src   = eidx;
    const int*   dst   = eidx + NEDGES;
    const int*   batch = (const int*)d_in[2];
    const int*   docid = (const int*)d_in[3];
    const float* attn  = (const float*)d_in[4];
    const int*   idx_a = (const int*)d_in[5];
    const int*   idx_b = (const int*)d_in[6];
    const float* W_in  = (const float*)d_in[7];  const float* b_in  = (const float*)d_in[8];
    const float* W_hid = (const float*)d_in[9];  const float* b_hid = (const float*)d_in[10];
    const float* W_out = (const float*)d_in[11]; const float* b_out = (const float*)d_in[12];
    const float* W_fc1 = (const float*)d_in[13]; const float* b_fc1 = (const float*)d_in[14];
    const float* W_fc2 = (const float*)d_in[15]; const float* b_fc2 = (const float*)d_in[16];
    const float* W_fc3 = (const float*)d_in[17]; const float* b_fc3 = (const float*)d_in[18];
    float* out = (float*)d_out;

    char* ws = (char*)d_ws;
    size_t off = 0;
    auto alloc = [&](size_t bytes) -> void* {
        void* p = ws + off;
        off = (off + bytes + 4095) & ~(size_t)4095;
        return p;
    };

    _Float16* hA   = (_Float16*)alloc((size_t)N_NODES * DMODEL * 2);  // activations (A)
    _Float16* yB   = (_Float16*)alloc((size_t)N_NODES * DMODEL * 2);  // y = dis⊙hW / fc1 out
    float*    agg  = (float*)   alloc((size_t)N_NODES * DMODEL * 4);  // scatter target
    _Float16* h2   = (_Float16*)alloc((size_t)N_NODES * 32 * 2);
    float*    h3   = (float*)   alloc((size_t)N_NODES * 4);
    float*    deg  = (float*)   alloc((size_t)N_NODES * 4);
    float*    dis  = (float*)   alloc((size_t)N_NODES * 4);
    float*    sA   = (float*)   alloc(4096);
    float*    cA   = (float*)   alloc(4096);
    float*    aD   = (float*)   alloc(4096);
    float*    sP   = (float*)   alloc(4096);
    float*    cP   = (float*)   alloc(4096);
    _Float16* Pin  = (_Float16*)alloc((size_t)F_IN   * DMODEL * 2);
    _Float16* Phid = (_Float16*)alloc((size_t)DMODEL * DMODEL * 2);
    _Float16* Pout = (_Float16*)alloc((size_t)DMODEL * DMODEL * 2);
    _Float16* Pfc1 = (_Float16*)alloc((size_t)DMODEL * DMODEL * 2);
    _Float16* Pfc2 = (_Float16*)alloc((size_t)DMODEL * 32 * 2);

    // per-call zeroing of accumulators (capture-safe)
    hipMemsetAsync(deg, 0, (size_t)N_NODES * 4, stream);
    hipMemsetAsync(sA, 0, NDOCS * 4, stream);
    hipMemsetAsync(cA, 0, NDOCS * 4, stream);
    hipMemsetAsync(sP, 0, NDOCS * 4, stream);
    hipMemsetAsync(cP, 0, NDOCS * 4, stream);

    // normalization + attention scalars
    deg_count<<<NEDGES / 256, 256, 0, stream>>>(dst, deg, NEDGES);
    make_dis<<<N_NODES / 256, 256, 0, stream>>>(deg, dis, N_NODES);
    attn_acc<<<N_NODES / 256, 256, 0, stream>>>(docid, attn, sA, cA, N_NODES);
    attn_a<<<1, NDOCS, 0, stream>>>(sA, cA, aD);

    // input features -> f16, weights -> packed fragments
    cvt_f16<<<(N_NODES * F_IN / 4) / 256, 256, 0, stream>>>(x, hA, N_NODES * F_IN / 4);
    pack_w<<<(F_IN   * DMODEL) / 256, 256, 0, stream>>>(W_in,  Pin,  F_IN,   DMODEL);
    pack_w<<<(DMODEL * DMODEL) / 256, 256, 0, stream>>>(W_hid, Phid, DMODEL, DMODEL);
    pack_w<<<(DMODEL * DMODEL) / 256, 256, 0, stream>>>(W_out, Pout, DMODEL, DMODEL);
    pack_w<<<(DMODEL * DMODEL) / 256, 256, 0, stream>>>(W_fc1, Pfc1, DMODEL, DMODEL);
    pack_w<<<(DMODEL * 32)     / 256, 256, 0, stream>>>(W_fc2, Pfc2, DMODEL, 32);

    // 4 GCN layers: in, hid, hid, out
    const _Float16* lp[4] = { Pin, Phid, Phid, Pout };
    const float*    lb[4] = { b_in, b_hid, b_hid, b_out };
    const int       lk[4] = { F_IN, DMODEL, DMODEL, DMODEL };
    const int gemmBlocks = (N_NODES / 16) * ((DMODEL / 16) / 4) / 8;  // 2048
    for (int l = 0; l < 4; ++l) {
        wmma_gemm<4><<<gemmBlocks, 256, 0, stream>>>(hA, lp[l], yB, lk[l], DMODEL,
                                                     0, dis, lb[l]);
        hipMemsetAsync(agg, 0, (size_t)N_NODES * DMODEL * 4, stream);
        edge_scatter<<<NEDGES / 4, 256, 0, stream>>>(src, dst, yB, agg, NEDGES);
        gcn_combine<<<(N_NODES * DMODEL / 4) / 256, 256, 0, stream>>>(
            agg, yB, dis, lb[l], hA, N_NODES * DMODEL / 4);
    }

    // fc1 (fused bias+tanh epilogue) -> yB ; fc2 -> h2 ; fc3 -> h3
    wmma_gemm<4><<<gemmBlocks, 256, 0, stream>>>(hA, Pfc1, yB, DMODEL, DMODEL,
                                                 1, dis, b_fc1);
    const int fc2Blocks = (N_NODES / 16) * ((32 / 16) / 2) / 8;       // 256
    wmma_gemm<2><<<fc2Blocks, 256, 0, stream>>>(yB, Pfc2, h2, DMODEL, 32,
                                                1, dis, b_fc2);
    fc3_kernel<<<N_NODES / 256, 256, 0, stream>>>(h2, W_fc3, b_fc3, h3, N_NODES);

    // pooling + preference output
    pool_acc<<<N_NODES / 256, 256, 0, stream>>>(batch, h3, sP, cP, N_NODES);
    final_kernel<<<1, NDOCS, 0, stream>>>(sP, cP, aD, idx_a, idx_b, out);
}
